// ACCNet_30923764531485
// MI455X (gfx1250) — matrix operand order repaired and checked by reading, not compiled, not run on hardware
//
#include <hip/hip_runtime.h>
#include <hip/hip_bf16.h>
#include <stdint.h>

// ---------------------------------------------------------------------------
// MI455X (gfx1250) implementation.
//   Stage 0: regenerate f16 trig table (cos/sin rows, zero-padded) -> ws
//   Stage 1: DFT power spectrum via v_wmma_f32_16x16x32_f16  (per-batch GEMM)
//   Stage 2: smoothing / peaks / stable bitonic argsort / threefry fallback /
//            windowed-sinc taps                               -> ws
//   Stage 3: dynamic FIR as banded-B WMMA GEMM, NT stores     -> d_out
// ---------------------------------------------------------------------------

typedef _Float16 half8  __attribute__((ext_vector_type(8)));
typedef _Float16 half16 __attribute__((ext_vector_type(16)));
typedef float    float8 __attribute__((ext_vector_type(8)));

#define NBINS 249
#define PI_F  3.14159265358979323846f

// ws layout (bytes)
#define TRIG_OFF 0                              // _Float16[512*512]
#define FPOW_OFF (512*512*2)                    // float[1024*256]
#define TAPS_OFF (FPOW_OFF + 1024*256*4)        // float[1024*3*101]
// total ~2.75 MB

// ---------------------------------------------------------------------------
// Stage 0: trig table. trig[n*512 + k]: n<256 -> cos(2*pi*n*k/500),
// n>=256 -> sin(2*pi*(n-256)*k/500). Rows j==0 or j>249, and k>=500 are 0.
// Exact angle reduction via integer mod so large j*k stays accurate.
// ---------------------------------------------------------------------------
__global__ __launch_bounds__(256) void k_trig(_Float16* __restrict__ trig) {
  int idx = blockIdx.x * 256 + threadIdx.x;     // 0 .. 512*512-1
  int n = idx >> 9, k = idx & 511;
  int j = n & 255;
  float v = 0.f;
  if (j >= 1 && j <= 249 && k < 500) {
    int m = (j * k) % 500;
    float ang = (2.0f * PI_F / 500.0f) * (float)m;
    v = (n < 256) ? cosf(ang) : sinf(ang);
  }
  trig[idx] = (_Float16)v;
}

// ---------------------------------------------------------------------------
// Stage 1: per-batch DFT power.  One block per batch b.
// A = x[b] (32x500 f32 -> f16 in LDS, K padded to 512).
// B = trig table (512 cols: cos j | sin j), fragments loaded straight from L2.
// 8 waves cover 2 mtiles x 32 ntiles; 16 k-steps of WMMA per tile.
// Deterministic reduction: 8 disjoint slots per column, fixed-order sum.
// ---------------------------------------------------------------------------
__global__ __launch_bounds__(256) void k_dft(const float* __restrict__ x,
                                             const _Float16* __restrict__ trig,
                                             float* __restrict__ fpow) {
  __shared__ _Float16 Ax[32][520];              // 512 + 8 pad halves
  __shared__ float pwp[8][256];                 // slot-split partial powers
  const int b = blockIdx.x;
  const int t = threadIdx.x;

  const float* xb = x + (size_t)b * (32 * 500);
  for (int i = t; i < 32 * 512; i += 256) {
    int c = i >> 9, l = i & 511;
    Ax[c][l] = (_Float16)((l < 500) ? xb[c * 500 + l] : 0.f);
  }
  __syncthreads();

  const int wave = t >> 5, lane = t & 31;
  const int lane15 = lane & 15, hi = lane >> 4;
  const int koffA = hi << 4, koffB = hi << 4;

  for (int tt = wave; tt < 64; tt += 8) {
    const int mt = tt >> 5;                     // 0..1  (channel halves)
    const int nt = tt & 31;                     // 0..31 (16-col tiles)
    const int row  = mt * 16 + lane15;
    const int ncol = nt * 16 + lane15;
    float8 acc = {};
    for (int kt = 0; kt < 16; ++kt) {
      const _Float16* pa = &Ax[row][kt * 32 + koffA];
      half8 a0 = *(const half8*)pa;
      half8 a1 = *(const half8*)(pa + 8);
      half16 af;
#pragma unroll
      for (int i2 = 0; i2 < 8; ++i2) { af[i2] = a0[i2]; af[i2 + 8] = a1[i2]; }
      half16 bf = *(const half16*)(trig + (size_t)ncol * 512 + kt * 32 + koffB);
      acc = __builtin_amdgcn_wmma_f32_16x16x32_f16(false, af, false, bf,
                                                   (short)0, acc, false, false);
    }
    float s = 0.f;
#pragma unroll
    for (int r = 0; r < 8; ++r) s += acc[r] * acc[r];
    const int slot = (((mt << 1) | (ncol >> 8)) << 1) | hi;   // unique writer
    pwp[slot][ncol & 255] = s;
  }
  __syncthreads();
  float acc0 = 0.f;
#pragma unroll
  for (int s = 0; s < 8; ++s) acc0 += pwp[s][t];
  fpow[(size_t)b * 256 + t] = acc0 * (1.0f / 32.0f);
}

// ---------------------------------------------------------------------------
// threefry2x32 with key = jax.random.key(777) = (0, 777); 20 rounds.
// ---------------------------------------------------------------------------
__device__ __forceinline__ uint32_t rotl32(uint32_t v, int r) {
  return (v << r) | (v >> (32 - r));
}
__device__ inline void threefry_0_777(uint32_t x0, uint32_t x1,
                                      uint32_t& o0, uint32_t& o1) {
  const uint32_t ks0 = 0u, ks1 = 777u;
  const uint32_t ks2 = 0x1BD11BDAu ^ ks0 ^ ks1;
  uint32_t v0 = x0 + ks0, v1 = x1 + ks1;
#define TF_ROUND(r) { v0 += v1; v1 = rotl32(v1, (r)); v1 ^= v0; }
  TF_ROUND(13) TF_ROUND(15) TF_ROUND(26) TF_ROUND(6)
  v0 += ks1; v1 += ks2 + 1u;
  TF_ROUND(17) TF_ROUND(29) TF_ROUND(16) TF_ROUND(24)
  v0 += ks2; v1 += ks0 + 2u;
  TF_ROUND(13) TF_ROUND(15) TF_ROUND(26) TF_ROUND(6)
  v0 += ks0; v1 += ks1 + 3u;
  TF_ROUND(17) TF_ROUND(29) TF_ROUND(16) TF_ROUND(24)
  v0 += ks1; v1 += ks2 + 4u;
  TF_ROUND(13) TF_ROUND(15) TF_ROUND(26) TF_ROUND(6)
  v0 += ks2; v1 += ks0 + 5u;
#undef TF_ROUND
  o0 = v0; o1 = v1;
}

// ---------------------------------------------------------------------------
// Stage 2: per-batch smoothing, peak pick, taps. One block per batch.
// Replicates the JAX g[2]->g[1] clamp (kernel [g0,g1,g1]) and stable
// descending argsort (key = ord(sm)<<32 | (0xFFFFFFFF - idx)).
// ---------------------------------------------------------------------------
__global__ __launch_bounds__(256) void k_peaks(const float* __restrict__ fpow,
                                               float* __restrict__ taps) {
  __shared__ float pw[256];
  __shared__ float sm[256];
  __shared__ unsigned long long key[256];
  __shared__ short ord[256];
  __shared__ unsigned char pk[256];
  __shared__ unsigned char vld[256];
  __shared__ float blo[3], bhi[3];
  const int b = blockIdx.x, t = threadIdx.x;

  pw[t] = fpow[(size_t)b * 256 + t];
  __syncthreads();

  const float g1u = expf(-0.125f);              // exp(-k^2/(2*sigma^2)), k=1
  const float G0 = 1.f / (1.f + g1u);
  const float G1 = g1u / (1.f + g1u);
  if (t < NBINS) {
    float fpm = (t >= 1)   ? pw[t]     : 0.f;   // fp[i]
    float fp0 = pw[t + 1];                      // fp[i+1]
    float fpp = (t <= 247) ? pw[t + 2] : 0.f;   // fp[i+2]
    sm[t] = G0 * fpm + G1 * fp0 + G1 * fpp;
  } else {
    sm[t] = 0.f;
  }
  __syncthreads();

  if (t < NBINS) {
    bool p = (t >= 1) && (t <= 247) && (sm[t] > sm[t - 1]) && (sm[t + 1] < sm[t]);
    pk[t] = p ? 1 : 0;
    uint32_t u = __float_as_uint(sm[t]);
    uint32_t o = (u & 0x80000000u) ? ~u : (u | 0x80000000u);  // order-preserving
    key[t] = ((unsigned long long)o << 32) |
             (unsigned long long)(0xFFFFFFFFu - (uint32_t)t); // stable ties
  } else {
    pk[t] = 0; key[t] = 0ull;                   // pads sort to the bottom
  }
  __syncthreads();

  // bitonic sort, descending, 256 keys
  for (int k2 = 2; k2 <= 256; k2 <<= 1) {
    for (int j2 = k2 >> 1; j2 > 0; j2 >>= 1) {
      int p2 = t ^ j2;
      if (p2 > t) {
        unsigned long long a = key[t], c = key[p2];
        bool doSwap = ((t & k2) == 0) ? (a < c) : (a > c);
        if (doSwap) { key[t] = c; key[p2] = a; }
      }
      __syncthreads();
    }
  }
  ord[t] = (short)(0xFFFFFFFFu - (uint32_t)(key[t] & 0xFFFFFFFFull));
  __syncthreads();

  // valid[i]: is a peak AND >12 bins (>3 Hz) from every stronger sorted bin
  bool v = false;
  if (t < NBINS) {
    int oi = ord[t];
    if (pk[oi]) {
      v = true;
      for (int j3 = 0; j3 < t; ++j3) {
        int d = oi - ord[j3]; d = (d < 0) ? -d : d;
        if (d <= 12) { v = false; break; }
      }
    }
  }
  vld[t] = v ? 1 : 0;
  __syncthreads();

  if (t == 0) {
    float pf[3];
    int cnt = 0;
    for (int i = 0; i < NBINS && cnt < 3; ++i)
      if (vld[i]) pf[cnt++] = (float)(ord[i] + 1) * 0.25f;
    for (int kk = cnt; kk < 3; ++kk) {          // jax randint(key(777)) fallback
      uint32_t p = (uint32_t)(b * 3 + kk);
      uint32_t o0, o1;
      threefry_0_777(p, p + 3072u, o0, o1);     // bits[0], bits[1]
      uint32_t ridx = ((o0 % 249u) * 160u + (o1 % 249u)) % 249u; // 160=(2^16%249)^2%249
      pf[kk] = (float)(ridx + 1u) * 0.25f;
    }
    float t0 = fminf(pf[0], pf[1]), t1 = fmaxf(pf[0], pf[1]);
    float m0 = fminf(t0, pf[2]);
    float m2 = fmaxf(t1, pf[2]);
    float m1 = fmaxf(t0, fminf(t1, pf[2]));
    float c1 = (m1 - m0) * 0.5f + m0;
    float c2 = (m2 - m1) * 0.5f + m1;
    blo[0] = 0.5f; bhi[0] = c1;
    blo[1] = c1;   bhi[1] = c2;
    blo[2] = c2;   bhi[2] = 45.f;
  }
  __syncthreads();

  for (int i = t; i < 3 * 101; i += 256) {
    int f = i / 101, n = i - f * 101;
    float lo = blo[f] * (1.f / 125.f);
    float hi = bhi[f] * (1.f / 125.f);
    float nn = (float)(n - 50);
    float tap = (n == 50) ? (hi - lo)
                          : (sinf(PI_F * nn * hi) - sinf(PI_F * nn * lo)) / (PI_F * nn);
    float win = 0.5f * (1.f - cosf(2.f * PI_F * (float)n / 101.f));
    taps[((size_t)b * 3 + f) * 101 + n] = tap * win;
  }
}

// ---------------------------------------------------------------------------
// Stage 3: dynamic FIR as WMMA GEMM.  out[b, r=f*32+c, l] with source channel
// r/3.  A'[m,q] = xpad[ch, 16t+q], B'[q,i] = taps[q-i] (banded), K=128 (4
// WMMA k-steps).  One block per batch; 6 mtiles x 32 ntiles; NT stores.
// ---------------------------------------------------------------------------
__global__ __launch_bounds__(256) void k_conv(const float* __restrict__ x,
                                              const float* __restrict__ taps,
                                              float* __restrict__ out) {
  __shared__ _Float16 Xp[32][640];              // xpad: l' = l + 50, zeros pad
  __shared__ _Float16 BT[3][16][128];           // B' transposed: [f][i][q]
  const int b = blockIdx.x, t = threadIdx.x;

  const float* xb = x + (size_t)b * (32 * 500);
  for (int i = t; i < 32 * 640; i += 256) {
    int c = i / 640, l = i - c * 640;
    int gl = l - 50;
    Xp[c][l] = (_Float16)((gl >= 0 && gl < 500) ? xb[c * 500 + gl] : 0.f);
  }
  const float* tb = taps + (size_t)b * (3 * 101);
  for (int i = t; i < 3 * 16 * 128; i += 256) {
    int f = i >> 11;
    int r = i & 2047;
    int n = r >> 7;
    int q = r & 127;
    int d = q - n;
    BT[f][n][q] = (_Float16)((d >= 0 && d < 101) ? tb[f * 101 + d] : 0.f);
  }
  __syncthreads();

  const int wave = t >> 5, lane = t & 31;
  const int lane15 = lane & 15, hi = lane >> 4;
  const int koffA = hi << 4, koffB = hi << 4;

  for (int tt = wave; tt < 192; tt += 8) {
    const int mt = tt >> 5;                     // 0..5
    const int nt = tt & 31;                     // 0..31
    const int f  = mt >> 1;
    const int row = mt * 16 + lane15;           // output row r = f*32 + c
    const int ch  = row / 3;                    // (32f + c) // 3
    const int l0  = nt * 16;
    float8 acc = {};
#pragma unroll
    for (int kt = 0; kt < 4; ++kt) {
      const _Float16* pa = &Xp[ch][l0 + kt * 32 + koffA];
      half8 a0 = *(const half8*)pa;
      half8 a1 = *(const half8*)(pa + 8);
      half16 af;
#pragma unroll
      for (int i2 = 0; i2 < 8; ++i2) { af[i2] = a0[i2]; af[i2 + 8] = a1[i2]; }
      half16 bf = *(const half16*)&BT[f][lane15][kt * 32 + koffB];
      acc = __builtin_amdgcn_wmma_f32_16x16x32_f16(false, af, false, bf,
                                                   (short)0, acc, false, false);
    }
    const int l = l0 + lane15;
    if (l < 500) {
#pragma unroll
      for (int rr = 0; rr < 8; ++rr) {
        int r = mt * 16 + rr + (hi << 3);
        __builtin_nontemporal_store(acc[rr], &out[((size_t)b * 96 + r) * 500 + l]);
      }
    }
  }
}

// ---------------------------------------------------------------------------
extern "C" void kernel_launch(void* const* d_in, const int* in_sizes, int n_in,
                              void* d_out, int out_size, void* d_ws, size_t ws_size,
                              hipStream_t stream) {
  const float* x = (const float*)d_in[0];       // (32768, 500) f32
  float* out = (float*)d_out;                   // (1024, 3, 32, 500) f32
  char* ws = (char*)d_ws;
  _Float16* trig = (_Float16*)(ws + TRIG_OFF);
  float* fpow    = (float*)(ws + FPOW_OFF);
  float* taps    = (float*)(ws + TAPS_OFF);

  k_trig <<<1024, 256, 0, stream>>>(trig);
  k_dft  <<<1024, 256, 0, stream>>>(x, trig, fpow);
  k_peaks<<<1024, 256, 0, stream>>>(fpow, taps);
  k_conv <<<1024, 256, 0, stream>>>(x, taps, out);
}
